// DiffnapsNet_42030549959310
// MI455X (gfx1250) — 16-bit and 8-bit matrix operand order repaired and checked by reading, compiled, not run on hardware
//
#include <hip/hip_runtime.h>
#include <hip/hip_bf16.h>


typedef __attribute__((ext_vector_type(8)))  int      v8i;
typedef __attribute__((ext_vector_type(8)))  float    v8f;
typedef __attribute__((ext_vector_type(16))) _Float16 v16h;
typedef int v4i_vs __attribute__((vector_size(16)));

#define TILE_K     64
#define ROW_STRIDE 80
#define BLK_M      128
#define BLK_N      256

#define GLB_AS __attribute__((address_space(1)))
#define LDS_AS __attribute__((address_space(3)))

#if defined(__HIP_DEVICE_COMPILE__) && __has_builtin(__builtin_amdgcn_global_load_async_to_lds_b128)
#define HAVE_ASYNC_BUILTIN 1
#else
#define HAVE_ASYNC_BUILTIN 0
#endif

__device__ __forceinline__ void async_copy_b128(const void* g, void* l) {
#if HAVE_ASYNC_BUILTIN
    __builtin_amdgcn_global_load_async_to_lds_b128(
        (GLB_AS v4i_vs*)(uintptr_t)g,
        (LDS_AS v4i_vs*)(unsigned int)(uintptr_t)l,
        0, 0);
#else
    unsigned int       lofs = (unsigned int)(uintptr_t)l;
    unsigned long long gadr = (unsigned long long)(uintptr_t)g;
    asm volatile("global_load_async_to_lds_b128 %0, %1, off"
                 :: "v"(lofs), "v"(gadr) : "memory");
#endif
}

__device__ __forceinline__ void wait_async0() {
#if defined(__HIP_DEVICE_COMPILE__) && __has_builtin(__builtin_amdgcn_s_wait_asynccnt)
    __builtin_amdgcn_s_wait_asynccnt(0);
#elif defined(__HIP_DEVICE_COMPILE__)
    asm volatile("s_wait_asynccnt 0" ::: "memory");
#endif
}

__device__ __forceinline__ v8i make_v8i_2(int2 a, int2 b, int2 c, int2 d) {
    v8i r;
    r[0] = a.x; r[1] = a.y; r[2] = b.x; r[3] = b.y;
    r[4] = c.x; r[5] = c.y; r[6] = d.x; r[7] = d.y;
    return r;
}

__device__ __forceinline__ v8i make_v8i_4(int4 a, int4 b) {
    v8i r;
    r[0] = a.x; r[1] = a.y; r[2] = a.z; r[3] = a.w;
    r[4] = b.x; r[5] = b.y; r[6] = b.z; r[7] = b.w;
    return r;
}

__global__ void binarize_pack_kernel(const float* __restrict__ in,
                                     unsigned char* __restrict__ out, int n4) {
    int i = blockIdx.x * blockDim.x + threadIdx.x;
    if (i < n4) {
        const float4 v = ((const float4*)in)[i];
        unsigned int r = (v.x > 0.5f ? 0x00000001u : 0u)
                       | (v.y > 0.5f ? 0x00000100u : 0u)
                       | (v.z > 0.5f ? 0x00010000u : 0u)
                       | (v.w > 0.5f ? 0x01000000u : 0u);
        ((unsigned int*)out)[i] = r;
    }
}

__global__ void cvt_f16_kernel(const float* __restrict__ in,
                               _Float16* __restrict__ out, int n) {
    int i = blockIdx.x * blockDim.x + threadIdx.x;
    if (i < n) out[i] = (_Float16)in[i];
}

__global__ __launch_bounds__(256)
void transpose_u8_kernel(const unsigned char* __restrict__ in,
                         unsigned char* __restrict__ out, int R, int C) {
    __shared__ unsigned char tile[64 * ROW_STRIDE];
    const int t  = threadIdx.x;
    const int r0 = blockIdx.y * 64;
    const int c0 = blockIdx.x * 64;
    {
        int lr = t >> 2;
        int lc = (t & 3) * 16;
        int4 v = *(const int4*)(in + (size_t)(r0 + lr) * C + c0 + lc);
        *(int4*)(&tile[lr * ROW_STRIDE + lc]) = v;
    }
    __syncthreads();
    {
        int oc  = t >> 2;
        int orr = (t & 3) * 16;
        union { unsigned char b[16]; int4 v; } u;
#pragma unroll
        for (int i = 0; i < 16; ++i) u.b[i] = tile[(orr + i) * ROW_STRIDE + oc];
        *(int4*)(out + (size_t)(c0 + oc) * R + r0 + orr) = u.v;
    }
}

__global__ __launch_bounds__(256)
void gemm_bin_kernel(const unsigned char* __restrict__ A,
                     const unsigned char* __restrict__ Bm,
                     const float* __restrict__ bias,
                     float* __restrict__ outF,
                     unsigned char* __restrict__ outB,
                     _Float16* __restrict__ outH,
                     int M, int N, int K) {
    __shared__ unsigned char smA[2][BLK_M * ROW_STRIDE];
    __shared__ unsigned char smB[2][BLK_N * ROW_STRIDE];

    const int tid  = threadIdx.x;
    const int lane = tid & 31;
    const int wid  = tid >> 5;
    const int wm   = wid >> 2;
    const int wn   = wid & 3;
    const int m0   = blockIdx.y * BLK_M;
    const int n0   = blockIdx.x * BLK_N;
    const int half = lane >> 4;
    const int l16  = lane & 15;

    const v8i zero8 = {0, 0, 0, 0, 0, 0, 0, 0};
    v8i acc[4][4];
#pragma unroll
    for (int mt = 0; mt < 4; ++mt)
#pragma unroll
        for (int nt = 0; nt < 4; ++nt) acc[mt][nt] = zero8;

    const int frowA = tid >> 1;
    const int fcolA = (tid & 1) * 32;
    const unsigned char* gA = A  + (size_t)(m0 + frowA) * K + fcolA;
    const unsigned char* gB = Bm + (size_t)(n0 + tid) * K;

    auto fill = [&](int buf, int kt) {
        const unsigned char* pa = gA + kt * TILE_K;
        async_copy_b128(pa,      &smA[buf][frowA * ROW_STRIDE + fcolA]);
        async_copy_b128(pa + 16, &smA[buf][frowA * ROW_STRIDE + fcolA + 16]);
        const unsigned char* pb = gB + kt * TILE_K;
#pragma unroll
        for (int c = 0; c < 64; c += 16)
            async_copy_b128(pb + c, &smB[buf][tid * ROW_STRIDE + c]);
    };

    const int nk = K / TILE_K;
    fill(0, 0);

    for (int kt = 0; kt < nk; ++kt) {
        const int cur = kt & 1;
        wait_async0();
        __syncthreads();
        if (kt + 1 < nk) fill(cur ^ 1, kt + 1);
        if (kt + 2 < nk) {
            __builtin_prefetch(gA + (kt + 2) * TILE_K, 0, 1);
            __builtin_prefetch(gB + (kt + 2) * TILE_K, 0, 1);
        }

        v8i afrag[4];
#pragma unroll
        for (int mt = 0; mt < 4; ++mt) {
            const unsigned char* base =
                &smA[cur][(wm * 64 + mt * 16 + l16) * ROW_STRIDE];
            const int2* p = (const int2*)(base + half * 8);
            afrag[mt] = make_v8i_2(p[0], p[2], p[4], p[6]);
        }
        v8i bfrag[4];
#pragma unroll
        for (int nt = 0; nt < 4; ++nt) {
            const unsigned char* base =
                &smB[cur][(wn * 64 + nt * 16 + l16) * ROW_STRIDE];
            const int2* q = (const int2*)(base + half * 8);
            bfrag[nt] = make_v8i_2(q[0], q[2], q[4], q[6]);
        }

#pragma unroll
        for (int mt = 0; mt < 4; ++mt)
#pragma unroll
            for (int nt = 0; nt < 4; ++nt)
                acc[mt][nt] = __builtin_amdgcn_wmma_i32_16x16x64_iu8(
                    false, afrag[mt], false, bfrag[nt], acc[mt][nt],
                    false, false);
    }

#pragma unroll
    for (int nt = 0; nt < 4; ++nt) {
        const int n  = n0 + wn * 64 + nt * 16 + l16;
        const float bv = bias[n];
#pragma unroll
        for (int mt = 0; mt < 4; ++mt) {
#pragma unroll
            for (int r = 0; r < 8; ++r) {
                const int m = m0 + wm * 64 + mt * 16 + r + half * 8;
                const float hv = (float)acc[mt][nt][r] + bv;
                const float zf = hv > 1.0f ? 1.0f : 0.0f;
                const size_t idx = (size_t)m * N + n;
                outF[idx] = zf;
                if (outB) outB[idx] = (unsigned char)zf;
                if (outH) outH[idx] = (_Float16)zf;
            }
        }
    }
}

__global__ __launch_bounds__(256)
void classify_kernel(const _Float16* __restrict__ zh,
                     const _Float16* __restrict__ clf,
                     float* __restrict__ out, int Hdim, int Ldim) {
    const int tid   = threadIdx.x;
    const int lane  = tid & 31;
    const int wid   = tid >> 5;
    const int half  = lane >> 4;
    const int l16   = lane & 15;
    const int mbase = blockIdx.x * 128 + wid * 16;

    const v8f zero8f = {0.f, 0.f, 0.f, 0.f, 0.f, 0.f, 0.f, 0.f};
    v8f acc[8];
#pragma unroll
    for (int nt = 0; nt < 8; ++nt) acc[nt] = zero8f;

    const _Float16* arow = zh + (size_t)(mbase + l16) * Hdim;
    for (int kk = 0; kk < Hdim; kk += 32) {
        int4 a0 = *(const int4*)(arow + kk + half * 8);
        int4 a1 = *(const int4*)(arow + kk + 16 + half * 8);
        const v16h af = __builtin_bit_cast(v16h, make_v8i_4(a0, a1));
#pragma unroll
        for (int nt = 0; nt < 8; ++nt) {
            const _Float16* brow =
                clf + (size_t)(nt * 16 + l16) * Hdim + kk + half * 8;
            int4 b0 = *(const int4*)(brow);
            int4 b1 = *(const int4*)(brow + 16);
            const v16h bf = __builtin_bit_cast(v16h, make_v8i_4(b0, b1));
            acc[nt] = __builtin_amdgcn_wmma_f32_16x16x32_f16(
                false, af, false, bf, (short)0, acc[nt], false, false);
        }
    }

#pragma unroll
    for (int nt = 0; nt < 8; ++nt) {
        const int n = nt * 16 + l16;
#pragma unroll
        for (int r = 0; r < 8; ++r) {
            const int m = mbase + r + half * 8;
            out[(size_t)m * Ldim + n] = acc[nt][r];
        }
    }
}

extern "C" void kernel_launch(void* const* d_in, const int* in_sizes, int n_in,
                              void* d_out, int out_size, void* d_ws,
                              size_t ws_size, hipStream_t stream) {
    (void)in_sizes; (void)n_in; (void)out_size; (void)ws_size;

    constexpr int B = 4096, D = 8192, H = 4096, L = 128;

    const float* x      = (const float*)d_in[0];
    const float* enc_w  = (const float*)d_in[1];
    const float* bias0  = (const float*)d_in[2];
    const float* bias3  = (const float*)d_in[3];
    const float* clf_w  = (const float*)d_in[4];

    float* outP       = (float*)d_out;
    float* out_output = outP;
    float* out_cls    = out_output + (size_t)B * D;
    float* out_z      = out_cls + (size_t)B * L;

    unsigned char* ws  = (unsigned char*)d_ws;
    unsigned char* xb  = ws;
    unsigned char* wb  = xb  + (size_t)B * D;
    unsigned char* wbT = wb  + (size_t)H * D;
    unsigned char* zb  = wbT + (size_t)D * H;
    _Float16*      zh  = (_Float16*)(zb + (size_t)B * H);
    _Float16*      clfh = zh + (size_t)B * H;

    binarize_pack_kernel<<<(B * D / 4 + 255) / 256, 256, 0, stream>>>(x, xb, B * D / 4);
    binarize_pack_kernel<<<(H * D / 4 + 255) / 256, 256, 0, stream>>>(enc_w, wb, H * D / 4);
    transpose_u8_kernel<<<dim3(D / 64, H / 64), 256, 0, stream>>>(wb, wbT, H, D);
    cvt_f16_kernel<<<(L * H + 255) / 256, 256, 0, stream>>>(clf_w, clfh, L * H);

    gemm_bin_kernel<<<dim3(H / BLK_N, B / BLK_M), 256, 0, stream>>>(
        xb, wb, bias0, out_z, zb, zh, B, H, D);

    classify_kernel<<<B / 128, 256, 0, stream>>>(zh, clfh, out_cls, H, L);

    gemm_bin_kernel<<<dim3(D / BLK_N, B / BLK_M), 256, 0, stream>>>(
        zb, wbT, bias3, out_output, nullptr, nullptr, B, D, H);
}
